// EdgeRegGNN_32615981646453
// MI455X (gfx1250) — compile-verified, matrix operand order checked by reading, not run on hardware
//
#include <hip/hip_runtime.h>

// ---------------------------------------------------------------------------
// EdgeRegGNN for MI455X (gfx1250, wave32, WMMA).
//   deg -> dinv -> [x@W1 WMMA bf16] -> selfloop+scatter agg -> relu+b1 (bf16)
//       -> [h@W2 WMMA bf16, bf16 input] -> selfloop+scatter agg -> relu+b2
//       -> edge MLP concat(h[row],h[col]) @ Wl WMMA bf16 (+bl)
// Hidden activations are stored packed-bf16: consumers (WMMA A operands) take
// them with raw b128 loads, no per-use conversion VALU.
// ---------------------------------------------------------------------------

typedef __attribute__((ext_vector_type(16))) __bf16 v16bf;
typedef __attribute__((ext_vector_type(8)))  float  v8f;

#define FIN   128   // input node features
#define HDIM  64    // hidden channels
#define FEDGE 8     // edge output features

__device__ __forceinline__ unsigned short f2bf_bits(float f) {
  unsigned int u = __builtin_bit_cast(unsigned int, f);
  u = (u + 0x7FFFu + ((u >> 16) & 1u)) >> 16;   // round-to-nearest-even
  return (unsigned short)u;
}
__device__ __forceinline__ unsigned int pk_bf2(float lo, float hi) {
  return (unsigned int)f2bf_bits(lo) | ((unsigned int)f2bf_bits(hi) << 16);
}

// Inverse of the WMMA bf16 operand K-mapping (ISA 7.12.2):
// element i of lane (half*16 + row) holds K = (i<8?0:16) + half*8 + (i&7)
__device__ __forceinline__ void kpos(int kw, int& half, int& i) {
  half = (kw >> 3) & 1;
  i    = ((kw & 16) ? 8 : 0) + (kw & 7);
}

// ---------------------------------------------------------------------------
// degree / normalization
// ---------------------------------------------------------------------------
__global__ void deg_init_kernel(float* __restrict__ deg, int n) {
  int v = blockIdx.x * blockDim.x + threadIdx.x;
  if (v < n) deg[v] = 1.0f;                       // self-loop
}

__global__ void deg_count_kernel(const int* __restrict__ col,
                                 float* __restrict__ deg, int e) {
  int t = blockIdx.x * blockDim.x + threadIdx.x;
  if (t < e) atomicAdd(&deg[col[t]], 1.0f);
}

__global__ void dinv_kernel(const float* __restrict__ deg,
                            float* __restrict__ dinv, int n) {
  int v = blockIdx.x * blockDim.x + threadIdx.x;
  if (v < n) dinv[v] = rsqrtf(fmaxf(deg[v], 1e-12f));
}

// ---------------------------------------------------------------------------
// GEMM (f32 input): out[n_rows x 64] = A[n_rows x 128] @ W[128 x 64]
// 256 threads = 8 waves; each wave: 16 rows x 64 cols (4 WMMA N-tiles).
// Operands staged in LDS pre-swizzled into WMMA register layout.
// ---------------------------------------------------------------------------
__global__ __launch_bounds__(256) void gemm_f32in_kernel(
    const float* __restrict__ A, const float* __restrict__ W,
    float* __restrict__ out, int n_rows) {
  constexpr int KB = FIN / 32;                    // 4 k-blocks
  __shared__ v16bf sA[8 * KB * 32];               // [wave][kb][lane]  32KB
  __shared__ v16bf sB[KB * 4 * 32];               // [kb][nt][lane]    16KB

  const int tid  = threadIdx.x;
  const int wave = tid >> 5;
  const int lane = tid & 31;
  const int row0 = blockIdx.x * 128;

  for (int p = tid; p < (FIN / 2) * HDIM; p += 256) {   // W: even-k pairs
    int k2 = p / HDIM, n = p % HDIM;
    int k = 2 * k2;
    float lo = W[k * HDIM + n], hi = W[(k + 1) * HDIM + n];
    int kb = k >> 5, half, i;
    kpos(k & 31, half, i);
    int vec = (kb * 4 + (n >> 4)) * 32 + half * 16 + (n & 15);
    ((unsigned int*)&sB[vec])[i >> 1] = pk_bf2(lo, hi);
  }
  for (int p = tid; p < 128 * (FIN / 2); p += 256) {    // A tile
    int r = p / (FIN / 2), k2 = p % (FIN / 2);
    int k = 2 * k2;
    int rr = (row0 + r < n_rows) ? (row0 + r) : 0;
    float2 q = ((const float2*)(A + (size_t)rr * FIN))[k2];
    int kb = k >> 5, half, i;
    kpos(k & 31, half, i);
    int vec = ((r >> 4) * KB + kb) * 32 + half * 16 + (r & 15);
    ((unsigned int*)&sA[vec])[i >> 1] = pk_bf2(q.x, q.y);
  }
  __syncthreads();

  const v16bf* pA = sA + wave * KB * 32;
  v8f acc[4] = {};
#pragma unroll
  for (int kb = 0; kb < KB; ++kb) {
    v16bf a = pA[kb * 32 + lane];
#pragma unroll
    for (int nt = 0; nt < 4; ++nt) {
      v16bf b = sB[(kb * 4 + nt) * 32 + lane];
      acc[nt] = __builtin_amdgcn_wmma_f32_16x16x32_bf16(
          false, a, false, b, (short)0, acc[nt], false, false);
    }
  }

  const int half = lane >> 4, ml = lane & 15;     // C/D: n=ml, m=half*8+j
#pragma unroll
  for (int nt = 0; nt < 4; ++nt)
#pragma unroll
    for (int j = 0; j < 8; ++j) {
      int m = row0 + wave * 16 + half * 8 + j;
      if (m < n_rows) out[(size_t)m * HDIM + nt * 16 + ml] = acc[nt][j];
    }
}

// ---------------------------------------------------------------------------
// GEMM (packed-bf16 input): out[n_rows x 64] = H[n_rows x 64] @ W[64 x 64]
// A staging is a pure dword reposition -> no conversion VALU at all.
// ---------------------------------------------------------------------------
__global__ __launch_bounds__(256) void gemm_bf16in_kernel(
    const unsigned int* __restrict__ Hp,   // packed bf16 pairs, 32 per row
    const float* __restrict__ W,
    float* __restrict__ out, int n_rows) {
  constexpr int KB = HDIM / 32;                   // 2 k-blocks
  __shared__ v16bf sA[8 * KB * 32];               // 16KB
  __shared__ v16bf sB[KB * 4 * 32];               // 8KB

  const int tid  = threadIdx.x;
  const int wave = tid >> 5;
  const int lane = tid & 31;
  const int row0 = blockIdx.x * 128;

  for (int p = tid; p < (HDIM / 2) * HDIM; p += 256) {  // W: even-k pairs
    int k2 = p / HDIM, n = p % HDIM;
    int k = 2 * k2;
    float lo = W[k * HDIM + n], hi = W[(k + 1) * HDIM + n];
    int kb = k >> 5, half, i;
    kpos(k & 31, half, i);
    int vec = (kb * 4 + (n >> 4)) * 32 + half * 16 + (n & 15);
    ((unsigned int*)&sB[vec])[i >> 1] = pk_bf2(lo, hi);
  }
  for (int p = tid; p < 128 * (HDIM / 2); p += 256) {   // A: raw dword move
    int r = p >> 5, k2 = p & 31;
    int k = 2 * k2;
    int rr = (row0 + r < n_rows) ? (row0 + r) : 0;
    unsigned int q = Hp[(size_t)rr * (HDIM / 2) + k2];
    int kb = k >> 5, half, i;
    kpos(k & 31, half, i);
    int vec = ((r >> 4) * KB + kb) * 32 + half * 16 + (r & 15);
    ((unsigned int*)&sA[vec])[i >> 1] = q;
  }
  __syncthreads();

  const v16bf* pA = sA + wave * KB * 32;
  v8f acc[4] = {};
#pragma unroll
  for (int kb = 0; kb < KB; ++kb) {
    v16bf a = pA[kb * 32 + lane];
#pragma unroll
    for (int nt = 0; nt < 4; ++nt) {
      v16bf b = sB[(kb * 4 + nt) * 32 + lane];
      acc[nt] = __builtin_amdgcn_wmma_f32_16x16x32_bf16(
          false, a, false, b, (short)0, acc[nt], false, false);
    }
  }

  const int half = lane >> 4, ml = lane & 15;
#pragma unroll
  for (int nt = 0; nt < 4; ++nt)
#pragma unroll
    for (int j = 0; j < 8; ++j) {
      int m = row0 + wave * 16 + half * 8 + j;
      if (m < n_rows) out[(size_t)m * HDIM + nt * 16 + ml] = acc[nt][j];
    }
}

// ---------------------------------------------------------------------------
// Aggregation (f32): agg = selfloop + sum_{edges} norm * h[row]
// ---------------------------------------------------------------------------
__global__ void self_init_kernel(const float* __restrict__ hpre,
                                 const float* __restrict__ dinv,
                                 float* __restrict__ agg, int n) {
  int t = blockIdx.x * blockDim.x + threadIdx.x;
  if (t < n * HDIM) {
    int v = t >> 6;
    float d = dinv[v];
    agg[t] = d * d * hpre[t];
  }
}

__global__ void edge_agg_kernel(const int* __restrict__ row,
                                const int* __restrict__ col,
                                const float* __restrict__ hpre,
                                const float* __restrict__ dinv,
                                float* __restrict__ agg, int e) {
  int t = blockIdx.x * blockDim.x + threadIdx.x;
  if (t < e * HDIM) {
    int ed = t >> 6;
    int f  = t & 63;
    int r = row[ed], c = col[ed];
    float nv = dinv[r] * dinv[c];
    atomicAdd(&agg[c * HDIM + f], nv * hpre[r * HDIM + f]);
  }
}

// relu(agg + b) quantized to packed bf16 (the only consumers are WMMA A ops)
__global__ void bias_relu_pack_kernel(const float* __restrict__ agg,
                                      const float* __restrict__ bias,
                                      unsigned int* __restrict__ hp, int n) {
  int t = blockIdx.x * blockDim.x + threadIdx.x;   // one packed pair each
  if (t < n * (HDIM / 2)) {
    int f2 = (t & 31) * 2;
    float lo = fmaxf(agg[2 * t]     + bias[f2],     0.0f);
    float hi = fmaxf(agg[2 * t + 1] + bias[f2 + 1], 0.0f);
    hp[t] = pk_bf2(lo, hi);
  }
}

// ---------------------------------------------------------------------------
// Edge MLP: out[e,8] = [h[row], h[col]] @ Wl[128x8] + bl   (h is packed bf16)
// 256 threads = 8 waves; 16 edges/wave; Wl padded to 128x16 pre-swizzled in
// LDS. A operand = two b128 loads per K-step, no conversion.
// ---------------------------------------------------------------------------
__global__ __launch_bounds__(256) void edge_mlp_kernel(
    const unsigned short* __restrict__ h,   // bf16, row stride 64
    const int* __restrict__ row, const int* __restrict__ col,
    const float* __restrict__ Wl, const float* __restrict__ bl,
    float* __restrict__ out, int e) {
  __shared__ v16bf sB[4 * 32];                    // [kb][lane], zero-padded

  const int tid = threadIdx.x;
  for (int p = tid; p < 64 * 16; p += 256) {      // k-pairs x 16 cols
    int k2 = p >> 4, n = p & 15;
    int k = 2 * k2;
    float lo = (n < FEDGE) ? Wl[k * FEDGE + n] : 0.0f;
    float hi = (n < FEDGE) ? Wl[(k + 1) * FEDGE + n] : 0.0f;
    int kb = k >> 5, half, i;
    kpos(k & 31, half, i);
    ((unsigned int*)&sB[kb * 32 + half * 16 + n])[i >> 1] = pk_bf2(lo, hi);
  }
  __syncthreads();

  const int wave = tid >> 5;
  const int lane = tid & 31;
  const int half = lane >> 4;
  const int ml   = lane & 15;
  const int e0   = blockIdx.x * 128 + wave * 16;

  int ea = e0 + ml;
  int ec = (ea < e) ? ea : 0;
  const unsigned short* pr = h + (size_t)row[ec] * HDIM;
  const unsigned short* pc = h + (size_t)col[ec] * HDIM;

  v8f acc = {};
#pragma unroll
  for (int kb = 0; kb < 4; ++kb) {
    const unsigned short* src = (kb < 2) ? pr : pc;
    const int kbase = (kb & 1) * 32;
    // elements 0..7  <- src[kbase +      half*8 + 0..7]   (16B, aligned)
    // elements 8..15 <- src[kbase + 16 + half*8 + 0..7]   (16B, aligned)
    uint4 q0 = *(const uint4*)(src + kbase + half * 8);
    uint4 q1 = *(const uint4*)(src + kbase + 16 + half * 8);
    v16bf a;
    unsigned int* au = (unsigned int*)&a;
    au[0] = q0.x; au[1] = q0.y; au[2] = q0.z; au[3] = q0.w;
    au[4] = q1.x; au[5] = q1.y; au[6] = q1.z; au[7] = q1.w;
    v16bf b = sB[kb * 32 + lane];
    acc = __builtin_amdgcn_wmma_f32_16x16x32_bf16(
        false, a, false, b, (short)0, acc, false, false);
  }

  if (ml < FEDGE) {                               // C/D: n=ml, m=half*8+j
    float bias = bl[ml];
#pragma unroll
    for (int j = 0; j < 8; ++j) {
      int ee = e0 + half * 8 + j;
      if (ee < e) out[(size_t)ee * FEDGE + ml] = acc[j] + bias;
    }
  }
}

// ---------------------------------------------------------------------------
// Launch
// ---------------------------------------------------------------------------
extern "C" void kernel_launch(void* const* d_in, const int* in_sizes, int n_in,
                              void* d_out, int out_size, void* d_ws, size_t ws_size,
                              hipStream_t stream) {
  const float* x   = (const float*)d_in[0];
  const int*   ei  = (const int*)d_in[1];
  const float* W1  = (const float*)d_in[2];
  const float* b1  = (const float*)d_in[3];
  const float* W2  = (const float*)d_in[4];
  const float* b2  = (const float*)d_in[5];
  const float* Wl  = (const float*)d_in[6];
  const float* bl  = (const float*)d_in[7];
  float* out = (float*)d_out;

  const int N = in_sizes[0] / FIN;
  const int E = in_sizes[1] / 2;
  const int* row = ei;         // edge_index[0]
  const int* col = ei + E;     // edge_index[1]

  // Workspace partition, each region 16B-aligned:
  //   deg[N] f32 | dinv[N] f32 | buf0[N*64] f32 | buf1[N*64] f32 | hbf[N*32] u32
  const size_t Na = ((size_t)N + 3) & ~(size_t)3;
  float* deg  = (float*)d_ws;
  float* dinv = deg  + Na;
  float* buf0 = dinv + Na;                 // GEMM output (pre-aggregation)
  float* buf1 = buf0 + (size_t)N * HDIM;   // aggregation accumulator
  unsigned int* hbf = (unsigned int*)(buf1 + (size_t)N * HDIM); // packed bf16 h
  (void)ws_size; (void)n_in; (void)out_size;

  const int T = 256;
  dim3 blk(T);

  // --- normalization ---
  deg_init_kernel<<<(N + T - 1) / T, blk, 0, stream>>>(deg, N);
  deg_count_kernel<<<(E + T - 1) / T, blk, 0, stream>>>(col, deg, E);
  dinv_kernel<<<(N + T - 1) / T, blk, 0, stream>>>(deg, dinv, N);

  const int gemm_blocks = (N + 127) / 128;
  const int nf = N * HDIM, ef = E * HDIM, np = N * (HDIM / 2);

  // --- layer 1: h1 = relu(agg(x@W1) + b1), packed bf16 ---
  gemm_f32in_kernel<<<gemm_blocks, 256, 0, stream>>>(x, W1, buf0, N);
  self_init_kernel<<<(nf + T - 1) / T, blk, 0, stream>>>(buf0, dinv, buf1, N);
  edge_agg_kernel<<<(ef + T - 1) / T, blk, 0, stream>>>(row, col, buf0, dinv, buf1, E);
  bias_relu_pack_kernel<<<(np + T - 1) / T, blk, 0, stream>>>(buf1, b1, hbf, N);

  // --- layer 2: h2 = relu(agg(h1@W2) + b2), packed bf16 ---
  gemm_bf16in_kernel<<<gemm_blocks, 256, 0, stream>>>(hbf, W2, buf0, N);
  self_init_kernel<<<(nf + T - 1) / T, blk, 0, stream>>>(buf0, dinv, buf1, N);
  edge_agg_kernel<<<(ef + T - 1) / T, blk, 0, stream>>>(row, col, buf0, dinv, buf1, E);
  bias_relu_pack_kernel<<<(np + T - 1) / T, blk, 0, stream>>>(buf1, b2, hbf, N);

  // --- edge MLP ---
  edge_mlp_kernel<<<(E + 127) / 128, 256, 0, stream>>>(
      (const unsigned short*)hbf, row, col, Wl, bl, out, E);
}
